// rnn_net_4_constitutive_fixed_sig_general_67980742361270
// MI455X (gfx1250) — matrix operand order repaired and checked by reading, not compiled, not run on hardware
//
#include <hip/hip_runtime.h>

// Problem sizes (fixed by the reference)
#define BB 4096
#define SS 512
#define II 6
#define HH 64
#define OO 6

typedef __attribute__((ext_vector_type(16))) __bf16 v16bf;
typedef __attribute__((ext_vector_type(8)))  __bf16 v8bf;
typedef __attribute__((ext_vector_type(8)))  float  v8f;

__device__ __forceinline__ v8f zero8() {
    v8f r;
#pragma unroll
    for (int i = 0; i < 8; ++i) r[i] = 0.f;
    return r;
}
__device__ __forceinline__ v16bf zero16bf() {
    v16bf r;
#pragma unroll
    for (int i = 0; i < 16; ++i) r[i] = (__bf16)0.f;
    return r;
}

// Hardware reciprocal (v_rcp_f32, ~1ulp) instead of IEEE div expansion.
__device__ __forceinline__ float fast_rcp(float x) {
    return __builtin_amdgcn_rcpf(x);
}
__device__ __forceinline__ float sigmoidf_(float x) {
    return fast_rcp(1.f + __expf(-x));
}
__device__ __forceinline__ float tanhf_(float x) {
    // tanh(x) = 1 - 2/(exp(2x)+1)  (saturates correctly at +/-inf)
    float e = __expf(2.f * x);
    return 1.f - 2.f * fast_rcp(e + 1.f);
}

// D = A*B + C, bf16 inputs, f32 accum, 16x16x32
__device__ __forceinline__ v8f wmma_bf16(v16bf a, v16bf b, v8f c) {
    return __builtin_amdgcn_wmma_f32_16x16x32_bf16(
        /*neg_a=*/false, a, /*neg_b=*/false, b,
        /*c_mod=*/(short)0, c, /*reuse_a=*/false, /*reuse_b=*/false);
}

// A fragment (16x32, M=row, K=hidden) from a row-major [16][64] bf16 LDS buffer.
// A layout: lanes 0-15 hold halves = K(0..7,16..23); lanes 16-31 hold K(8..15,24..31).
__device__ __forceinline__ v16bf load_a_frag(const __bf16* sh, int ln, int hi, int kf) {
    const __bf16* p = sh + ln * HH + kf * 32 + hi * 8;
    v8bf lo = *(const v8bf*)(p);
    v8bf hh = *(const v8bf*)(p + 16);
    v16bf a;
#pragma unroll
    for (int t = 0; t < 8; ++t) { a[t] = lo[t]; a[8 + t] = hh[t]; }
    return a;
}

// Pre-packed B fragment: LDS buffer laid out [32 lanes][16 halves] per fragment.
// Uniform across lanes: two contiguous ds_load_b128, no EXEC games, no zero-fill.
__device__ __forceinline__ v16bf load_fragB(const __bf16* frag, int lane) {
    const __bf16* q = frag + lane * 16;
    v8bf lo = *(const v8bf*)(q);
    v8bf hh = *(const v8bf*)(q + 8);
    v16bf b;
#pragma unroll
    for (int t = 0; t < 8; ++t) { b[t] = lo[t]; b[8 + t] = hh[t]; }
    return b;
}

__global__ __launch_bounds__(128) void gru3_fused_kernel(
    const float* __restrict__ x,
    const float* __restrict__ xmean, const float* __restrict__ xstd,
    const float* __restrict__ ymin, const float* __restrict__ ymax,
    const float* __restrict__ wih0, const float* __restrict__ whh0,
    const float* __restrict__ bih0, const float* __restrict__ bhh0,
    const float* __restrict__ wih1, const float* __restrict__ whh1,
    const float* __restrict__ bih1, const float* __restrict__ bhh1,
    const float* __restrict__ wih2, const float* __restrict__ whh2,
    const float* __restrict__ bih2, const float* __restrict__ bhh2,
    const float* __restrict__ wlin, const float* __restrict__ blin,
    float* __restrict__ y)
{
    // LDS: weights pre-packed as per-lane WMMA B fragments ([frag][lane][half]),
    // plus per-cell hidden state (row-major [16][64], this is the D->A transpose).
    __shared__ __bf16 sh_whhB[3][24][512];  // 72 KB : frag f = tile*2+kf, tile 0..11
    __shared__ __bf16 sh_wihB[3][12][512];  // 36 KB : frag = tile, K padded 6->32
    __shared__ __bf16 sh_wlinB[2][512];     //  2 KB : frag = kf
    __shared__ __bf16 sh_h[3][16 * 64];     //  6 KB

    const int tid  = threadIdx.x;
    const int wave = tid >> 5;
    const int lane = tid & 31;
    const int hi   = lane >> 4;
    const int ln   = lane & 15;
    const int base = blockIdx.x * 16;   // 16 batch rows per workgroup

    // ---- prologue: pack weights (f32 -> bf16) into fragment-layout LDS ----
    // B layout: lane(hi2,ln2) half h -> element B[k][n] with
//   k = 32*kf + 8*hi2 + (h&7) + 16*(h>>3),  n = ln2 (column = gate/output index 16*tile+ln2)
    for (int i = tid; i < 3 * 24 * 512; i += 128) {
        int c2 = i / (24 * 512); int r = i % (24 * 512);
        int f = r >> 9; int e = r & 511;
        int lane2 = e >> 4, half = e & 15;
        int hi2 = lane2 >> 4, ln2 = lane2 & 15;
        int tile = f >> 1, kf = f & 1;
        int k = kf * 32 + hi2 * 8 + (half & 7) + ((half >> 3) << 4);
        int g = tile * 16 + ln2;
        const float* w = (c2 == 0) ? whh0 : ((c2 == 1) ? whh1 : whh2);
        sh_whhB[c2][f][e] = (__bf16)w[g * HH + k];
    }
    for (int i = tid; i < 3 * 12 * 512; i += 128) {
        int c2 = i / (12 * 512); int r = i % (12 * 512);
        int f = r >> 9; int e = r & 511;
        int lane2 = e >> 4, half = e & 15;
        int hi2 = lane2 >> 4, ln2 = lane2 & 15;
        int k = hi2 * 16 + half;              // kf == 0 only
        int g = f * 16 + ln2;
        const float* wi = (c2 == 0) ? wih0 : ((c2 == 1) ? wih1 : wih2);
        sh_wihB[c2][f][e] = (__bf16)((k < II) ? wi[g * II + k] : 0.f);
    }
    for (int i = tid; i < 2 * 512; i += 128) {
        int kf = i >> 9; int e = i & 511;
        int lane2 = e >> 4, half = e & 15;
        int hi2 = lane2 >> 4, ln2 = lane2 & 15;
        int k = kf * 32 + hi2 * 8 + (half & 7) + ((half >> 3) << 4);
        sh_wlinB[kf][e] = (__bf16)((ln2 < OO) ? wlin[ln2 * HH + k] : 0.f);
    }
    for (int i = tid; i < 3 * 16 * 64; i += 128)
        sh_h[i / (16 * 64)][i % (16 * 64)] = (__bf16)0.f;
    __syncthreads();

    // ---- per-wave persistent state ----
    const int c = (wave < 3) ? wave : 0;
    const float* bihp = (c == 0) ? bih0 : ((c == 1) ? bih1 : bih2);
    const float* bhhp = (c == 0) ? bhh0 : ((c == 1) ? bhh1 : bhh2);
    float brz[4], bzz[4], bihn[4], bhhn[4];
#pragma unroll
    for (int j = 0; j < 4; ++j) {
        int col = j * 16 + ln;
        brz[j]  = bihp[col] + bhhp[col];            // r-gate: biases always summed
        bzz[j]  = bihp[64 + col] + bhhp[64 + col];  // z-gate: biases always summed
        bihn[j] = bihp[128 + col];                  // n-gate: kept separate (r * (gh+bhh))
        bhhn[j] = bhhp[128 + col];
    }
    float xmr[II], xsr[II];
#pragma unroll
    for (int i = 0; i < II; ++i) { xmr[i] = xmean[i]; xsr[i] = 1.f / xstd[i]; }

    v8f hD[4];  // f32 hidden state, D layout: row = 8*hi+v, col = 16*j+ln
#pragma unroll
    for (int j = 0; j < 4; ++j) hD[j] = zero8();

    const __bf16* myh = sh_h[c];
    const __bf16 (*mywhhB)[512] = sh_whhB[c];
    const __bf16 (*mywihB)[512] = sh_wihB[c];

    // head-wave state (wave 3)
    float hmn = 0.f, hrg = 0.f, hbl = 0.f;
    if (ln < OO) { hmn = ymin[ln]; hrg = ymax[ln] - ymin[ln]; hbl = blin[ln]; }
    v16bf wB0 = load_fragB(sh_wlinB[0], lane);
    v16bf wB1 = load_fragB(sh_wlinB[1], lane);

    // ---- recurrence over time ----
#pragma unroll 1
    for (int s = 0; s < SS; ++s) {
        v8f hnew[4];
        if (wave < 3) {
            // A fragments of h_s (from LDS) and normalized x_s (from global)
            v16bf hA0 = load_a_frag(myh, ln, hi, 0);
            v16bf hA1 = load_a_frag(myh, ln, hi, 1);
            v16bf xA = zero16bf();
            if (hi == 0) {
                const float* xp = x + ((size_t)(base + ln) * SS + s) * II;
#pragma unroll
                for (int i = 0; i < II; ++i)
                    xA[i] = (__bf16)((xp[i] - xmr[i]) * xsr[i]);
            }
#pragma unroll
            for (int j = 0; j < 4; ++j) {
                // fetch all 9 B fragments up front (uniform ds_load_b128 pairs)
                v16bf br0 = load_fragB(mywhhB[(j)     * 2 + 0], lane);
                v16bf br1 = load_fragB(mywhhB[(j)     * 2 + 1], lane);
                v16bf bz0 = load_fragB(mywhhB[(4 + j) * 2 + 0], lane);
                v16bf bz1 = load_fragB(mywhhB[(4 + j) * 2 + 1], lane);
                v16bf bn0 = load_fragB(mywhhB[(8 + j) * 2 + 0], lane);
                v16bf bn1 = load_fragB(mywhhB[(8 + j) * 2 + 1], lane);
                v16bf bir = load_fragB(mywihB[j],     lane);
                v16bf biz = load_fragB(mywihB[4 + j], lane);
                v16bf bin = load_fragB(mywihB[8 + j], lane);
                // three independent accumulation chains, interleaved
                v8f ar = wmma_bf16(hA0, br0, zero8());
                v8f az = wmma_bf16(hA0, bz0, zero8());
                v8f an = wmma_bf16(hA0, bn0, zero8());
                ar = wmma_bf16(hA1, br1, ar);
                az = wmma_bf16(hA1, bz1, az);
                an = wmma_bf16(hA1, bn1, an);
                ar = wmma_bf16(xA, bir, ar);
                az = wmma_bf16(xA, biz, az);
                v8f gin = wmma_bf16(xA, bin, zero8());
#pragma unroll
                for (int v = 0; v < 8; ++v) {
                    float r  = sigmoidf_(ar[v] + brz[j]);
                    float z  = sigmoidf_(az[v] + bzz[j]);
                    float nn = tanhf_(gin[v] + bihn[j] + r * (an[v] + bhhn[j]));
                    hnew[j][v] = (1.f - z) * nn + z * hD[j][v];
                }
            }
        }
        __syncthreads();   // head done reading h_s; cell compute done
        if (wave < 3) {
            // publish h_{s+1} as bf16 row-major (this is the D->A transpose)
#pragma unroll
            for (int j = 0; j < 4; ++j) {
                hD[j] = hnew[j];
                int col = j * 16 + ln;
#pragma unroll
                for (int v = 0; v < 8; ++v)
                    sh_h[c][(hi * 8 + v) * HH + col] = (__bf16)hnew[j][v];
            }
        }
        __syncthreads();   // h_{s+1} visible to everyone
        if (wave == 3) {
            // head: logits = (h1+h2+h3) @ w_lin^T, then sigmoid range-scale
            v8f acc = zero8();
#pragma unroll
            for (int kf = 0; kf < 2; ++kf) {
                // vector bf16 adds -> v_pk_add_bf16 candidates
                const __bf16* p0 = sh_h[0] + ln * HH + kf * 32 + hi * 8;
                const __bf16* p1 = sh_h[1] + ln * HH + kf * 32 + hi * 8;
                const __bf16* p2 = sh_h[2] + ln * HH + kf * 32 + hi * 8;
                v8bf lo = *(const v8bf*)(p0)      + *(const v8bf*)(p1)      + *(const v8bf*)(p2);
                v8bf hh = *(const v8bf*)(p0 + 16) + *(const v8bf*)(p1 + 16) + *(const v8bf*)(p2 + 16);
                v16bf sa;
#pragma unroll
                for (int t = 0; t < 8; ++t) { sa[t] = lo[t]; sa[8 + t] = hh[t]; }
                acc = wmma_bf16(sa, (kf == 0) ? wB0 : wB1, acc);
            }
            if (ln < OO) {
#pragma unroll
                for (int v = 0; v < 8; ++v) {
                    float l  = acc[v] + hbl;
                    float yy = sigmoidf_(l) * hrg + hmn;
                    y[((size_t)(base + hi * 8 + v) * SS + s) * OO + ln] = yy;
                }
            }
        }
    }
}

extern "C" void kernel_launch(void* const* d_in, const int* in_sizes, int n_in,
                              void* d_out, int out_size, void* d_ws, size_t ws_size,
                              hipStream_t stream) {
    (void)in_sizes; (void)n_in; (void)out_size; (void)d_ws; (void)ws_size;
    const float* x    = (const float*)d_in[0];
    const float* xm   = (const float*)d_in[1];
    const float* xs   = (const float*)d_in[2];
    const float* ymin = (const float*)d_in[3];
    const float* ymax = (const float*)d_in[4];
    const float* wih1 = (const float*)d_in[5];
    const float* whh1 = (const float*)d_in[6];
    const float* bih1 = (const float*)d_in[7];
    const float* bhh1 = (const float*)d_in[8];
    const float* wih2 = (const float*)d_in[9];
    const float* whh2 = (const float*)d_in[10];
    const float* bih2 = (const float*)d_in[11];
    const float* bhh2 = (const float*)d_in[12];
    const float* wih3 = (const float*)d_in[13];
    const float* whh3 = (const float*)d_in[14];
    const float* bih3 = (const float*)d_in[15];
    const float* bhh3 = (const float*)d_in[16];
    const float* wlin = (const float*)d_in[17];
    const float* blin = (const float*)d_in[18];
    float* y = (float*)d_out;

    dim3 grid(BB / 16);
    dim3 block(128);
    gru3_fused_kernel<<<grid, block, 0, stream>>>(
        x, xm, xs, ymin, ymax,
        wih1, whh1, bih1, bhh1,
        wih2, whh2, bih2, bhh2,
        wih3, whh3, bih3, bhh3,
        wlin, blin, y);
}
